// BartCustomAttention_43774306680883
// MI455X (gfx1250) — matrix operand order repaired and checked by reading, not compiled, not run on hardware
//
#include <hip/hip_runtime.h>
#include <hip/hip_bf16.h>
#include <stdint.h>

// ---------------------------------------------------------------------------
// BART custom attention with relation embeddings, gfx1250 (MI455X, wave32).
// All GEMMs on v_wmma_f32_16x16x32_bf16. Relation-V term folded into a
// 41-bin per-row histogram (K_REL_SCALE==0 kills the score-bias term).
// Workspace: 28 MB.
// ---------------------------------------------------------------------------

typedef __bf16 bf16_t;
typedef __attribute__((ext_vector_type(16))) __bf16 bf16x16;
typedef __attribute__((ext_vector_type(8)))  float  f32x8;

#define T_SEQ   1024
#define NHEAD   16
#define DHEAD   64
#define BB      2
#define DMODEL  1024
#define NRELP1  41

union FragBF { uint4 u[2]; bf16x16 v; };

__device__ __forceinline__ bf16_t f2bf(float f) {
  union { float f; uint32_t u; } x; x.f = f;
  uint32_t r = x.u + 0x7FFFu + ((x.u >> 16) & 1u);   // round-to-nearest-even
  uint16_t h = (uint16_t)(r >> 16);
  return __builtin_bit_cast(bf16_t, h);
}

__device__ __forceinline__ f32x8 wmma_bf16(bf16x16 a, bf16x16 b, f32x8 c) {
  // (neg_a, A, neg_b, B, c_mod, C, reuse_a, reuse_b)
  return __builtin_amdgcn_wmma_f32_16x16x32_bf16(false, a, false, b, (short)0, c,
                                                 false, false);
}

// ---------------------------------------------------------------------------
// f32 -> bf16 elementwise conversion
// ---------------------------------------------------------------------------
__global__ void __launch_bounds__(256)
cvt_bf16_kernel(const float* __restrict__ in, bf16_t* __restrict__ out, int n) {
  int i = blockIdx.x * 256 + threadIdx.x;
  if (i < n) out[i] = f2bf(in[i]);
}

// ---------------------------------------------------------------------------
// C = A[M,K] @ W[N,K]^T  (+bias)*scale, bf16 WMMA, f32 accumulate.
// Block: 256 threads = 8 waves; wave tile = 16(M) x 32(N); block = 64x64.
// mode 0: q  -> bf16 [B,H,T,dh], scale 1/8
// mode 1: k  -> bf16 [B,H,T,dh]
// mode 2: v  -> bf16 [B,H,dh,T]   (transposed for contiguous P*V B-fragments)
// mode 3: o  -> f32  [M,N] (d_out)
// ---------------------------------------------------------------------------
__global__ void __launch_bounds__(256)
gemm_bf16_kernel(const bf16_t* __restrict__ A, const bf16_t* __restrict__ W,
                 const float* __restrict__ bias,
                 bf16_t* __restrict__ out_bf, float* __restrict__ out_f,
                 int M, int N, int K, int mode, float scale)
{
  const int tid  = threadIdx.x;
  const int lane = tid & 31, wave = tid >> 5;
  const int lr = lane & 15, lh = lane >> 4;
  const int mi = wave & 3,  ni = wave >> 2;
  const int Mb = blockIdx.y * 64 + mi * 16;
  const int Nb = blockIdx.x * 64 + ni * 32;

  f32x8 acc0 = {}, acc1 = {};
  const bf16_t* ap  = A + (size_t)(Mb + lr) * K;        // A row per lane (M = lr)
  const bf16_t* wp0 = W + (size_t)(Nb + lr) * K;        // B col per lane (N = lr)
  const bf16_t* wp1 = W + (size_t)(Nb + 16 + lr) * K;

  for (int kb = 0; kb < K; kb += 32) {
    FragBF a, b0, b1;
    // A 16x32 bf16: lanes<16 hold K 0..15, lanes>=16 hold K 16..31
    a.u[0] = *(const uint4*)(ap + kb + lh * 16);
    a.u[1] = *(const uint4*)(ap + kb + lh * 16 + 8);
    // B 32x16 bf16: lanes<16 hold K 0..15, lanes>=16 hold K 16..31 (N = lane%16)
    b0.u[0] = *(const uint4*)(wp0 + kb + lh * 16);
    b0.u[1] = *(const uint4*)(wp0 + kb + lh * 16 + 8);
    b1.u[0] = *(const uint4*)(wp1 + kb + lh * 16);
    b1.u[1] = *(const uint4*)(wp1 + kb + lh * 16 + 8);
    if (kb + 32 < K) {                      // global_prefetch_b8 of next K-slice
      __builtin_prefetch(ap + kb + 32, 0, 0);
      __builtin_prefetch(wp0 + kb + 32, 0, 0);
      __builtin_prefetch(wp1 + kb + 32, 0, 0);
    }
    acc0 = wmma_bf16(a.v, b0.v, acc0);
    acc1 = wmma_bf16(a.v, b1.v, acc1);
  }

  // C layout: reg r, lane -> (m = r + 8*(lane>=16), n = lane%16)
  for (int j = 0; j < 2; ++j) {
    const f32x8 acc = j ? acc1 : acc0;
    const int n = Nb + j * 16 + lr;
    const float bn = bias[n];
    for (int r = 0; r < 8; ++r) {
      const int m = Mb + r + 8 * lh;
      const float val = (acc[r] + bn) * scale;
      if (mode == 3) {
        out_f[(size_t)m * N + n] = val;
      } else {
        const int b = m / T_SEQ, t = m % T_SEQ;
        const int h = n >> 6,    d = n & 63;
        if (mode == 2)
          out_bf[(((size_t)(b * NHEAD + h) * DHEAD) + d) * T_SEQ + t] = f2bf(val);
        else
          out_bf[(((size_t)(b * NHEAD + h) * T_SEQ) + t) * DHEAD + d] = f2bf(val);
      }
    }
  }
}

// ---------------------------------------------------------------------------
// Attention + relation-V histogram.
// Block: 128 threads = 4 waves; each wave owns 16 t-rows; block covers 64 rows
// of one (b,h). Two-pass online softmax, scores never touch HBM.
// ---------------------------------------------------------------------------
__global__ void __launch_bounds__(128)
attn_rel_kernel(const bf16_t* __restrict__ qbf, const bf16_t* __restrict__ kbf,
                const bf16_t* __restrict__ vT,  const int* __restrict__ relI,
                const float* __restrict__ rel_emb, bf16_t* __restrict__ obf)
{
  __shared__ float hist[64][48];                 // 41 bins (+pad) per t-row
  __shared__ float relS[NRELP1][64];             // rel_emb staged in LDS
  __shared__ __align__(16) bf16_t Pst[4][16][32]; // per-wave P relayout buffer

  const int tid  = threadIdx.x;
  const int lane = tid & 31, wave = tid >> 5;
  const int lr = lane & 15, lh = lane >> 4;

  const int tb = blockIdx.x & 15;          // t-block within sequence
  const int bh = blockIdx.x >> 4;          // b*16 + h
  const int h  = bh & 15, b = bh >> 4;
  const int t0 = tb * 64;

  for (int i = tid; i < 64 * 48; i += 128)     ((float*)hist)[i] = 0.f;
  for (int i = tid; i < NRELP1 * 64; i += 128) ((float*)relS)[i] = rel_emb[i];
  __syncthreads();

  // q A-fragments (16 rows x 64 K) resident in registers for the whole kernel
  const int trow = t0 + wave * 16 + lr;
  const bf16_t* qp = qbf + ((size_t)bh * T_SEQ + trow) * DHEAD;
  FragBF qa0, qa1;
  qa0.u[0] = *(const uint4*)(qp + lh * 16);
  qa0.u[1] = *(const uint4*)(qp + lh * 16 + 8);
  qa1.u[0] = *(const uint4*)(qp + 32 + lh * 16);
  qa1.u[1] = *(const uint4*)(qp + 32 + lh * 16 + 8);

  const bf16_t* kbase = kbf + (size_t)bh * T_SEQ * DHEAD;

  float mrow[8], lrow[8];
  for (int r = 0; r < 8; ++r) { mrow[r] = -3.0e38f; lrow[r] = 0.f; }

  // ---- pass 1: online row-max + row-sumexp ----
  for (int sb = 0; sb < T_SEQ; sb += 16) {
    const bf16_t* kp = kbase + (size_t)(sb + lr) * DHEAD;  // B-frag: col = k-row s
    FragBF kb0, kb1;
    kb0.u[0] = *(const uint4*)(kp + lh * 16);
    kb0.u[1] = *(const uint4*)(kp + lh * 16 + 8);
    kb1.u[0] = *(const uint4*)(kp + 32 + lh * 16);
    kb1.u[1] = *(const uint4*)(kp + 32 + lh * 16 + 8);
    f32x8 S = {};
    S = wmma_bf16(qa0.v, kb0.v, S);
    S = wmma_bf16(qa1.v, kb1.v, S);
    for (int r = 0; r < 8; ++r) {
      float x = S[r];
      for (int off = 1; off < 16; off <<= 1) x = fmaxf(x, __shfl_xor(x, off, 32));
      const float mn = fmaxf(mrow[r], x);
      float e = __expf(S[r] - mn);
      for (int off = 1; off < 16; off <<= 1) e += __shfl_xor(e, off, 32);
      lrow[r] = lrow[r] * __expf(mrow[r] - mn) + e;
      mrow[r] = mn;
    }
  }
  float irow[8];
  for (int r = 0; r < 8; ++r) irow[r] = (lrow[r] > 0.f) ? 1.f / lrow[r] : 0.f;

  // ---- pass 2: P = softmax(S); histogram; O += P @ V ----
  f32x8 acc[4] = {};
  const bf16_t* vbase = vT + (size_t)bh * DHEAD * T_SEQ;
  const int* relrow = relI + (size_t)b * T_SEQ * T_SEQ;

  for (int sb = 0; sb < T_SEQ; sb += 32) {
    for (int half = 0; half < 2; ++half) {
      const int s0 = sb + half * 16;
      const bf16_t* kp = kbase + (size_t)(s0 + lr) * DHEAD;
      FragBF kb0, kb1;
      kb0.u[0] = *(const uint4*)(kp + lh * 16);
      kb0.u[1] = *(const uint4*)(kp + lh * 16 + 8);
      kb1.u[0] = *(const uint4*)(kp + 32 + lh * 16);
      kb1.u[1] = *(const uint4*)(kp + 32 + lh * 16 + 8);
      f32x8 S = {};
      S = wmma_bf16(qa0.v, kb0.v, S);
      S = wmma_bf16(qa1.v, kb1.v, S);
      const int scol = s0 + lr;
      for (int r = 0; r < 8; ++r) {
        const float p = __expf(S[r] - mrow[r]) * irow[r];
        const int tl = wave * 16 + r + 8 * lh;
        const int rk = relrow[(size_t)(t0 + tl) * T_SEQ + scol];
        atomicAdd(&hist[tl][rk], p);                 // ds_add_f32
        Pst[wave][r + 8 * lh][half * 16 + lr] = f2bf(p);
      }
    }
    __syncthreads();   // C-layout writes -> A-layout reads (relayout via LDS)
    FragBF pa;
    const bf16_t* pp = &Pst[wave][lr][0];
    pa.u[0] = *(const uint4*)(pp + lh * 16);
    pa.u[1] = *(const uint4*)(pp + lh * 16 + 8);
    for (int dt = 0; dt < 4; ++dt) {
      const bf16_t* vp = vbase + (size_t)(dt * 16 + lr) * T_SEQ + sb + lh * 16;
      FragBF vb;
      vb.u[0] = *(const uint4*)(vp);
      vb.u[1] = *(const uint4*)(vp + 8);
      acc[dt] = wmma_bf16(pa.v, vb.v, acc[dt]);
    }
    __syncthreads();   // protect Pst before next iteration overwrites
  }
  __syncthreads();

  // ---- epilogue: O += hist @ rel_emb  (rel 0 row is zero -> skip) ----
  for (int rk = 1; rk < NRELP1; ++rk) {
    const float re0 = relS[rk][lr],      re1 = relS[rk][16 + lr];
    const float re2 = relS[rk][32 + lr], re3 = relS[rk][48 + lr];
    for (int r = 0; r < 8; ++r) {
      const float hv = hist[wave * 16 + r + 8 * lh][rk];
      acc[0][r] += hv * re0; acc[1][r] += hv * re1;
      acc[2][r] += hv * re2; acc[3][r] += hv * re3;
    }
  }
  // store bf16 [B*T, D] for the output projection
  for (int dt = 0; dt < 4; ++dt)
    for (int r = 0; r < 8; ++r) {
      const int tl = wave * 16 + r + 8 * lh;
      const int d  = dt * 16 + lr;
      obf[(size_t)(b * T_SEQ + t0 + tl) * DMODEL + h * DHEAD + d] = f2bf(acc[dt][r]);
    }
}

// ---------------------------------------------------------------------------
extern "C" void kernel_launch(void* const* d_in, const int* in_sizes, int n_in,
                              void* d_out, int out_size, void* d_ws, size_t ws_size,
                              hipStream_t stream)
{
  const float* hs      = (const float*)d_in[0];
  const int*   relI    = (const int*)  d_in[1];
  const float* Wq      = (const float*)d_in[2];
  const float* bq      = (const float*)d_in[3];
  const float* Wk      = (const float*)d_in[4];
  const float* bk      = (const float*)d_in[5];
  const float* Wv      = (const float*)d_in[6];
  const float* bv      = (const float*)d_in[7];
  const float* Wo      = (const float*)d_in[8];
  const float* bo      = (const float*)d_in[9];
  const float* rel_emb = (const float*)d_in[10];

  char* ws = (char*)d_ws;                       // 28 MB total
  bf16_t* Xbf  = (bf16_t*)(ws);                 // [2048,1024]  4 MB
  bf16_t* Wqbf = (bf16_t*)(ws + ( 4u << 20));   // [1024,1024]  2 MB
  bf16_t* Wkbf = (bf16_t*)(ws + ( 6u << 20));
  bf16_t* Wvbf = (bf16_t*)(ws + ( 8u << 20));
  bf16_t* Wobf = (bf16_t*)(ws + (10u << 20));
  bf16_t* qbf  = (bf16_t*)(ws + (12u << 20));   // [B,H,T,64]   4 MB
  bf16_t* kbf  = (bf16_t*)(ws + (16u << 20));
  bf16_t* vTbf = (bf16_t*)(ws + (20u << 20));   // [B,H,64,T]
  bf16_t* obf  = (bf16_t*)(ws + (24u << 20));   // [2048,1024]

  const int M = BB * T_SEQ, N = DMODEL, K = DMODEL;

  cvt_bf16_kernel<<<dim3((M * K + 255) / 256), 256, 0, stream>>>(hs, Xbf, M * K);
  cvt_bf16_kernel<<<dim3((N * K + 255) / 256), 256, 0, stream>>>(Wq, Wqbf, N * K);
  cvt_bf16_kernel<<<dim3((N * K + 255) / 256), 256, 0, stream>>>(Wk, Wkbf, N * K);
  cvt_bf16_kernel<<<dim3((N * K + 255) / 256), 256, 0, stream>>>(Wv, Wvbf, N * K);
  cvt_bf16_kernel<<<dim3((N * K + 255) / 256), 256, 0, stream>>>(Wo, Wobf, N * K);

  dim3 g(N / 64, M / 64);
  gemm_bf16_kernel<<<g, 256, 0, stream>>>(Xbf, Wqbf, bq, qbf,  nullptr, M, N, K, 0, 0.125f);
  gemm_bf16_kernel<<<g, 256, 0, stream>>>(Xbf, Wkbf, bk, kbf,  nullptr, M, N, K, 1, 1.0f);
  gemm_bf16_kernel<<<g, 256, 0, stream>>>(Xbf, Wvbf, bv, vTbf, nullptr, M, N, K, 2, 1.0f);

  attn_rel_kernel<<<dim3(BB * NHEAD * (T_SEQ / 64)), 128, 0, stream>>>(
      qbf, kbf, vTbf, relI, rel_emb, obf);

  gemm_bf16_kernel<<<g, 256, 0, stream>>>(obf, Wobf, bo, nullptr, (float*)d_out,
                                          M, N, K, 3, 1.0f);
}
